// MSTANChurnClassifier_14448269984410
// MI455X (gfx1250) — matrix operand order repaired and checked by reading, not compiled, not run on hardware
//
#include <hip/hip_runtime.h>
#include <hip/hip_bf16.h>

// ---------------------------------------------------------------------------
// MSTAN churn classifier, CDNA5 (gfx1250) implementation.
// All dense contractions go through v_wmma_f32_16x16x32_f16 (wave32 WMMA,
// f16 inputs / f32 accumulate). fp32 master activations live in d_ws.
// GEMM tiles are staged to LDS with GLOBAL_LOAD_ASYNC_TO_LDS_B128 (ASYNCcnt)
// when the toolchain exposes the builtin; synchronous fallback otherwise.
// ---------------------------------------------------------------------------

typedef __attribute__((ext_vector_type(16))) _Float16     v16h;
typedef __attribute__((ext_vector_type(8)))  float        v8f;
typedef __attribute__((ext_vector_type(4)))  unsigned int v4u;
typedef __attribute__((ext_vector_type(4)))  int          v4i;

struct __attribute__((aligned(16))) F4 { float x, y, z, w; };

union FragH { v16h v; v4u u[2]; _Float16 h[16]; };

#define WMMA16(A, B, C) \
  __builtin_amdgcn_wmma_f32_16x16x32_f16(false, (A), false, (B), (short)0, (C), false, false)

#if defined(__gfx1250__) && \
    __has_builtin(__builtin_amdgcn_global_load_async_to_lds_b128) && \
    __has_builtin(__builtin_amdgcn_s_wait_asynccnt)
#define HAVE_ASYNC_LDS 1
#else
#define HAVE_ASYNC_LDS 0
#endif

#if HAVE_ASYNC_LDS
// Builtin signature (from hipcc diagnostic): param0 = addrspace(1) v4i*
// (global source), param1 = addrspace(3) v4i* (LDS dest), imm offset, imm cpol.
__device__ __forceinline__ void async_cp16(void* lds, const void* g) {
  __builtin_amdgcn_global_load_async_to_lds_b128(
      (__attribute__((address_space(1))) v4i*)(g),
      (__attribute__((address_space(3))) v4i*)(lds),
      /*offset=*/0, /*cpol=*/0);
}
#endif

__device__ __forceinline__ float gelu_f(float v) {
  return 0.5f * v * (1.0f + erff(v * 0.70710678118654752f));
}

// ---------------------------------------------------------------------------
// Generic fp32-in GEMM via f16 WMMA. Block tile 128(M) x 64(N), BK=32.
// 256 threads = 8 waves; wave w owns rows w*16..w*16+15 and all 64 cols.
// fp32 tiles live in LDS (async-copied); f16 conversion happens on the
// fragment build.  OOB M rows are redirected (not zeroed): a garbage A row m
// only pollutes C row m, which the guarded epilogue never stores.
// EPI: 0=bias  1=bias+gelu  2=bias+residual  3=qkv f16 pack  4=msproj(+PE->h2)
// ---------------------------------------------------------------------------
struct GemmP {
  const float* A; const float* Bm; float* Cout;
  const float* bias; const float* resid;
  int M, N, K, lda, ldb, ldc;
  _Float16* qkv_base;           // EPI 3
  const float* hres; float* h2; // EPI 4
};

template <int EPI>
__global__ void __launch_bounds__(256) gemm_wmma(GemmP p) {
  __shared__ __align__(16) float As32[128][36]; // [m][k] fp32, padded rows (144B)
  __shared__ __align__(16) float Bs32[32][68];  // [k][n] fp32, padded rows (272B)
  const int tid  = threadIdx.x;
  const int lane = tid & 31, wv = tid >> 5;
  const int ln = lane & 15, lh = lane >> 4;
  const int mBase = blockIdx.x * 128;
  const int nBase = blockIdx.y * 64;
  const int r0 = wv * 16;

  v8f acc[4] = {};

  for (int kb = 0; kb < p.K; kb += 32) {
    // ---- stage A: 128x32 fp32 (4 x b128 per thread) ----
#pragma unroll
    for (int i = 0; i < 4; ++i) {
      int v = tid + i * 256;
      int row = v >> 3, c4 = (v & 7) << 2;
      int grow = mBase + row;
      const float* src =
          p.A + (size_t)(grow < p.M ? grow : mBase) * p.lda + kb + c4;
#if HAVE_ASYNC_LDS
      async_cp16(&As32[row][c4], src);
#else
      *(F4*)&As32[row][c4] = *(const F4*)src;
#endif
    }
    // ---- stage B: 32x64 fp32 row-major (2 x b128 per thread) ----
#pragma unroll
    for (int i = 0; i < 2; ++i) {
      int v = tid + i * 256;
      int kr = v >> 4, c4 = (v & 15) << 2;
      const float* src = p.Bm + (size_t)(kb + kr) * p.ldb + nBase + c4;
#if HAVE_ASYNC_LDS
      async_cp16(&Bs32[kr][c4], src);
#else
      *(F4*)&Bs32[kr][c4] = *(const F4*)src;
#endif
    }
#if HAVE_ASYNC_LDS
    __builtin_amdgcn_s_wait_asynccnt(0); // this wave's copies done
#else
    if (kb + 32 < p.K) {
      int grow = mBase + (tid & 127);
      if (grow < p.M) __builtin_prefetch(p.A + (size_t)grow * p.lda + kb + 32);
    }
#endif
    __syncthreads();                     // all waves' copies visible

    // A fragment: lanes l / l+16 share row M=r0+ln; K split 0-15 / 16-31
    FragH a;
#pragma unroll
    for (int e = 0; e < 8; ++e) {
      a.h[e]     = (_Float16)As32[r0 + ln][lh * 16 + e];
      a.h[e + 8] = (_Float16)As32[r0 + ln][lh * 16 + 8 + e];
    }
#pragma unroll
    for (int f4 = 0; f4 < 4; ++f4) {
      // B fragment: col N=f4*16+ln fixed, 16 contiguous K per lane half
      FragH b;
#pragma unroll
      for (int e = 0; e < 16; ++e)
        b.h[e] = (_Float16)Bs32[lh * 16 + e][f4 * 16 + ln];
      acc[f4] = WMMA16(a.v, b.v, acc[f4]);
    }
    __syncthreads();
  }

  // ---- epilogue: row decomposition hoisted out of the N-tile loop ----
#pragma unroll
  for (int r = 0; r < 8; ++r) {
    int grow = mBase + r0 + r + lh * 8;
    if (grow >= p.M) continue;
    int rb = 0, rt = 0;
    if (EPI == 3) { rb = grow / 449; rt = grow % 449; }
    if (EPI == 4) { rb = grow / 448; rt = grow % 448; }
#pragma unroll
    for (int f4 = 0; f4 < 4; ++f4) {
      int gcol = nBase + f4 * 16 + ln;
      float v = acc[f4][r] + p.bias[gcol];
      if (EPI == 0) {
        p.Cout[(size_t)grow * p.ldc + gcol] = v;
      } else if (EPI == 1) {
        p.Cout[(size_t)grow * p.ldc + gcol] = gelu_f(v);
      } else if (EPI == 2) {
        p.Cout[(size_t)grow * p.ldc + gcol] = v + p.resid[(size_t)grow * p.ldc + gcol];
      } else if (EPI == 3) { // pack q/k/v as f16 [which][b][head][t][32]
        int which = gcol >> 8, head = (gcol & 255) >> 5, d = gcol & 31;
        size_t idx = (size_t)which * (size_t)(64 * 8 * 449 * 32) +
                     (((size_t)(rb * 8 + head)) * 449 + rt) * 32 + d;
        p.qkv_base[idx] = (_Float16)v;
      } else { // 4: ms-proj: + residual h + sinusoidal PE, scatter into h2[b, t+1]
        int i2 = gcol >> 1;
        float dv  = expf((float)(2 * i2) * (-9.210340371976184f / 256.f));
        float ang = (float)rt * dv;
        float pe  = (gcol & 1) ? cosf(ang) : sinf(ang);
        float o   = v + p.hres[(size_t)grow * 256 + gcol] + pe;
        p.h2[((size_t)rb * 449 + (rt + 1)) * 256 + gcol] = o;
      }
    }
  }
}

// ---------------------------------------------------------------------------
// Dilated causal conv as 3 shifted 256x256 WMMA GEMM taps, fused
// bias + exact GELU + BatchNorm epilogue.  conv_w layout: [O=256][I=256][K=3].
// Synchronous staging kept here: the causal shift needs zero-fill at t<0,
// which async LDS copies cannot synthesize.
// ---------------------------------------------------------------------------
struct ConvP {
  const float* H; const float* Wt; const float* bias;
  const float* bng; const float* bnb; const float* rm; const float* rv;
  float* out; int dil;
};

__global__ void __launch_bounds__(256) conv_wmma(ConvP p) {
  __shared__ __align__(16) _Float16 As[128][48];
  __shared__ __align__(16) _Float16 Bs[64][48];
  const int M = 64 * 448;
  const int tid  = threadIdx.x;
  const int lane = tid & 31, wv = tid >> 5;
  const int ln = lane & 15, lh = lane >> 4;
  const int mBase = blockIdx.x * 128;
  const int nBase = blockIdx.y * 64;
  const int r0 = wv * 16;

  v8f acc[4] = {};

  for (int tap = 0; tap < 3; ++tap) {
    int shift = (2 - tap) * p.dil;
    for (int kb = 0; kb < 256; kb += 32) {
#pragma unroll
      for (int i = 0; i < 4; ++i) {
        int v = tid + i * 256;
        int row = v >> 3, c4 = (v & 7) << 2;
        int grow = mBase + row;
        F4 f = {0.f, 0.f, 0.f, 0.f};
        if (grow < M) {
          int bi = grow / 448, t = grow % 448;
          int st = t - shift;              // causal shift, masked at batch edge
          if (st >= 0) f = *(const F4*)(p.H + ((size_t)bi * 448 + st) * 256 + kb + c4);
        }
        As[row][c4 + 0] = (_Float16)f.x; As[row][c4 + 1] = (_Float16)f.y;
        As[row][c4 + 2] = (_Float16)f.z; As[row][c4 + 3] = (_Float16)f.w;
      }
#pragma unroll
      for (int i = 0; i < 8; ++i) {
        int e = tid + i * 256;
        int kk = e & 31, n = e >> 5;
        Bs[n][kk] = (_Float16)p.Wt[(((size_t)(nBase + n)) * 256 + (kb + kk)) * 3 + tap];
      }
      __syncthreads();
      FragH a;
      a.u[0] = *(const v4u*)&As[r0 + ln][lh * 16];
      a.u[1] = *(const v4u*)&As[r0 + ln][lh * 16 + 8];
#pragma unroll
      for (int f4 = 0; f4 < 4; ++f4) {
        FragH b;
        b.u[0] = *(const v4u*)&Bs[f4 * 16 + ln][lh * 16];
        b.u[1] = *(const v4u*)&Bs[f4 * 16 + ln][lh * 16 + 8];
        acc[f4] = WMMA16(a.v, b.v, acc[f4]);
      }
      __syncthreads();
    }
  }

#pragma unroll
  for (int r = 0; r < 8; ++r) {
    int grow = mBase + r0 + r + lh * 8;
    if (grow >= M) continue;
#pragma unroll
    for (int f4 = 0; f4 < 4; ++f4) {
      int gcol = nBase + f4 * 16 + ln;
      float v = acc[f4][r] + p.bias[gcol];
      v = gelu_f(v);
      v = (v - p.rm[gcol]) * rsqrtf(p.rv[gcol] + 1e-5f) * p.bng[gcol] + p.bnb[gcol];
      p.out[(size_t)grow * 256 + gcol] = v;
    }
  }
}

// ---------------------------------------------------------------------------
// Attention: one wave per (b, head, 16-row q tile).  Q*K^T via a single
// 16x16x32 WMMA per key tile (hd=32), 16x449 logit strip + softmax in LDS,
// then P*V via WMMA with on-the-fly f32->f16 of the probabilities.
// ---------------------------------------------------------------------------
__global__ void __launch_bounds__(32)
attn_wmma(const _Float16* qf, const _Float16* kf, const _Float16* vf,
          const float* alpha, int layer, float* outO) {
  __shared__ float probs[16][480];
  const int TT = 449, HD = 32;
  const int lane = threadIdx.x;
  const int ln = lane & 15, lh = lane >> 4;
  const int qt = blockIdx.x;
  const int bh = blockIdx.y;
  const int hh = bh & 7;
  const int q0 = qt * 16;
  const size_t base = (size_t)bh * TT * HD;

  float aRaw  = alpha[layer * 8 + hh];
  float aBias = (aRaw > 20.f) ? aRaw : log1pf(expf(aRaw)); // softplus
  const float invs = 0.17677669529663689f;                 // 1/sqrt(32)

  int tq = q0 + ln; if (tq >= TT) tq = TT - 1;
  FragH aq;
  aq.u[0] = *(const v4u*)(qf + base + (size_t)tq * HD + lh * 8);
  aq.u[1] = *(const v4u*)(qf + base + (size_t)tq * HD + lh * 8 + 16);

  for (int jt = 0; jt < 29; ++jt) { // 29 key tiles cover 449
    int kb = jt * 16;
    int kc = kb + ln; int kcc = (kc < TT) ? kc : TT - 1;
    FragH bk;
    bk.u[0] = *(const v4u*)(kf + base + (size_t)kcc * HD + lh * 8);
    bk.u[1] = *(const v4u*)(kf + base + (size_t)kcc * HD + lh * 8 + 16);
    v8f c = {};
    c = WMMA16(aq.v, bk.v, c);
#pragma unroll
    for (int r = 0; r < 8; ++r) {
      int qrow = r + lh * 8;
      int qpos = q0 + qrow;
      int kpos = kb + ln;
      float val = -3.0e38f;
      if (qpos < TT && kpos < TT)
        val = c[r] * invs - aBias * fabsf((float)(qpos - kpos));
      probs[qrow][kb + ln] = val;
    }
  }
  __syncthreads();

  if (lane < 16) { // row softmax over the 449 real keys; zero the pad
    float m = -3.0e38f;
    for (int cI = 0; cI < TT; ++cI) m = fmaxf(m, probs[lane][cI]);
    float s = 0.f;
    for (int cI = 0; cI < TT; ++cI) {
      float e = expf(probs[lane][cI] - m);
      probs[lane][cI] = e; s += e;
    }
    float inv = 1.f / s;
    for (int cI = 0; cI < TT; ++cI) probs[lane][cI] *= inv;
    for (int cI = TT; cI < 480; ++cI) probs[lane][cI] = 0.f;
  }
  __syncthreads();

  v8f o0 = {}, o1 = {};
  for (int ck = 0; ck < 15; ++ck) { // 15 * 32 = 480 padded keys
    FragH ap;
    int kb0 = ck * 32 + lh * 16;
#pragma unroll
    for (int e = 0; e < 8; ++e) {
      ap.h[e]     = (_Float16)probs[ln][kb0 + e];
      ap.h[e + 8] = (_Float16)probs[ln][kb0 + 8 + e];
    }
    FragH bv0, bv1;
#pragma unroll
    for (int e = 0; e < 16; ++e) {
      int key = ck * 32 + lh * 16 + e;
      _Float16 x0 = (_Float16)0.f, x1 = (_Float16)0.f;
      if (key < TT) {
        x0 = vf[base + (size_t)key * HD + ln];
        x1 = vf[base + (size_t)key * HD + 16 + ln];
      }
      bv0.h[e] = x0; bv1.h[e] = x1;
    }
    o0 = WMMA16(ap.v, bv0.v, o0);
    o1 = WMMA16(ap.v, bv1.v, o1);
  }
  int b = bh >> 3;
#pragma unroll
  for (int r = 0; r < 8; ++r) {
    int t = q0 + r + lh * 8;
    if (t < TT) {
      size_t o = ((size_t)b * TT + t) * 256 + hh * 32;
      outO[o + ln]      = o0[r];
      outO[o + 16 + ln] = o1[r];
    }
  }
}

// --------------------------- VALU helper kernels ---------------------------

__global__ void __launch_bounds__(256)
ln_kernel(const float* x, const float* g, const float* b, float* y, int rows) {
  int wv = threadIdx.x >> 5, lane = threadIdx.x & 31;
  int row = blockIdx.x * 8 + wv;
  if (row >= rows) return;
  const float* xr = x + (size_t)row * 256;
  float s = 0.f, s2 = 0.f, vb[8];
#pragma unroll
  for (int i = 0; i < 8; ++i) { float v = xr[lane + i * 32]; vb[i] = v; s += v; s2 += v * v; }
#pragma unroll
  for (int o = 16; o > 0; o >>= 1) { s += __shfl_xor(s, o, 32); s2 += __shfl_xor(s2, o, 32); }
  float mean = s * (1.f / 256.f);
  float var  = s2 * (1.f / 256.f) - mean * mean;
  float rstd = rsqrtf(var + 1e-5f);
  float* yr = y + (size_t)row * 256;
#pragma unroll
  for (int i = 0; i < 8; ++i) {
    int c = lane + i * 32;
    yr[c] = (vb[i] - mean) * rstd * g[c] + b[c];
  }
}

__global__ void __launch_bounds__(256)
gate_fuse(const float* c0, const float* c1, const float* c2,
          const float* gw, const float* gb, float* fused, int rows) {
  int wv = threadIdx.x >> 5, lane = threadIdx.x & 31;
  int row = blockIdx.x * 8 + wv;
  if (row >= rows) return;
  const float* s0 = c0 + (size_t)row * 256;
  const float* s1 = c1 + (size_t)row * 256;
  const float* s2 = c2 + (size_t)row * 256;
  float l0 = 0.f, l1 = 0.f, l2 = 0.f;
#pragma unroll
  for (int i = 0; i < 8; ++i) {
    int d = lane + i * 32;
    float v0 = s0[d], v1 = s1[d], v2 = s2[d];
    l0 += v0 * gw[d * 3 + 0] + v1 * gw[(256 + d) * 3 + 0] + v2 * gw[(512 + d) * 3 + 0];
    l1 += v0 * gw[d * 3 + 1] + v1 * gw[(256 + d) * 3 + 1] + v2 * gw[(512 + d) * 3 + 1];
    l2 += v0 * gw[d * 3 + 2] + v1 * gw[(256 + d) * 3 + 2] + v2 * gw[(512 + d) * 3 + 2];
  }
#pragma unroll
  for (int o = 16; o > 0; o >>= 1) {
    l0 += __shfl_xor(l0, o, 32); l1 += __shfl_xor(l1, o, 32); l2 += __shfl_xor(l2, o, 32);
  }
  l0 += gb[0]; l1 += gb[1]; l2 += gb[2];
  float m = fmaxf(l0, fmaxf(l1, l2));
  float e0 = expf(l0 - m), e1 = expf(l1 - m), e2 = expf(l2 - m);
  float inv = 1.f / (e0 + e1 + e2);
  float g0 = e0 * inv, g1 = e1 * inv, g2 = e2 * inv;
  float* fr = fused + (size_t)row * 256;
#pragma unroll
  for (int i = 0; i < 8; ++i) {
    int d = lane + i * 32;
    fr[d] = g0 * s0[d] + g1 * s1[d] + g2 * s2[d];
  }
}

__global__ void cls_fill(float* h2, const float* cls) {
  h2[(size_t)blockIdx.x * 449 * 256 + threadIdx.x] = cls[threadIdx.x];
}

__global__ void __launch_bounds__(256)
cls_head(const float* n, const float* w1, const float* b1,
         const float* w2, const float* b2, float* out) {
  __shared__ float red[256];
  int b = blockIdx.x, j = threadIdx.x;
  const float* c = n + (size_t)b * 449 * 256; // row 0 of batch b
  float acc = 0.f;
  for (int i = 0; i < 256; ++i) acc += c[i] * w1[(size_t)i * 256 + j];
  float hid = gelu_f(acc + b1[j]);
  red[j] = hid * w2[j];
  __syncthreads();
  for (int s = 128; s > 0; s >>= 1) { if (j < s) red[j] += red[j + s]; __syncthreads(); }
  if (j == 0) out[b] = red[0] + b2[0];
}

// ------------------------------- launcher ----------------------------------

extern "C" void kernel_launch(void* const* d_in, const int* in_sizes, int n_in,
                              void* d_out, int out_size, void* d_ws, size_t ws_size,
                              hipStream_t stream) {
  const float* x      = (const float*)d_in[0];
  const float* ipw    = (const float*)d_in[1];
  const float* ipb    = (const float*)d_in[2];
  const float* conv_w = (const float*)d_in[3];
  const float* conv_b = (const float*)d_in[4];
  const float* bn_g   = (const float*)d_in[5];
  const float* bn_b   = (const float*)d_in[6];
  const float* bn_rm  = (const float*)d_in[7];
  const float* bn_rv  = (const float*)d_in[8];
  const float* gate_w = (const float*)d_in[9];
  const float* gate_b = (const float*)d_in[10];
  const float* msw    = (const float*)d_in[11];
  const float* msb    = (const float*)d_in[12];
  const float* cls_t  = (const float*)d_in[13];
  const float* qkv_w  = (const float*)d_in[14];
  const float* qkv_b  = (const float*)d_in[15];
  const float* out_w  = (const float*)d_in[16];
  const float* out_b  = (const float*)d_in[17];
  const float* alpha  = (const float*)d_in[18];
  const float* ln1_g  = (const float*)d_in[19];
  const float* ln1_b  = (const float*)d_in[20];
  const float* ln2_g  = (const float*)d_in[21];
  const float* ln2_b  = (const float*)d_in[22];
  const float* ffw1   = (const float*)d_in[23];
  const float* ffb1   = (const float*)d_in[24];
  const float* ffw2   = (const float*)d_in[25];
  const float* ffb2   = (const float*)d_in[26];
  const float* fin_g  = (const float*)d_in[27];
  const float* fin_b  = (const float*)d_in[28];
  const float* cw1    = (const float*)d_in[29];
  const float* cb1    = (const float*)d_in[30];
  const float* cw2    = (const float*)d_in[31];
  const float* cb2    = (const float*)d_in[32];
  float* out = (float*)d_out;

  const int Bn = 64, TT = 449;
  const size_t rowsA = (size_t)Bn * 448; // 28672
  const size_t rowsB = (size_t)Bn * TT;  // 28736

  // workspace layout (fp32): h2 | buf1 (LN-out / fused / attn-out) | scratch
  // scratch is reused for: h + 3 conv outs -> packed f16 qkv -> FFN mid.
  float* h2    = (float*)d_ws;
  float* buf1  = h2 + rowsB * 256;
  float* scr   = buf1 + rowsB * 256;       // 28736*1024 floats
  float* hbuf  = scr;
  float* cbuf0 = scr + rowsA * 256;
  float* cbuf1 = cbuf0 + rowsA * 256;
  float* cbuf2 = cbuf1 + rowsA * 256;
  _Float16* qkvh = (_Float16*)scr;
  float* ffnMid  = scr;

  dim3 blk(256);

  // input projection: [28672,64] @ [64,256]
  { GemmP p = {x, ipw, hbuf, ipb, nullptr,
               (int)rowsA, 256, 64, 64, 256, 256, nullptr, nullptr, nullptr};
    gemm_wmma<0><<<dim3(224, 4), blk, 0, stream>>>(p); }

  // multi-scale dilated convs (+bias+gelu+BN)
  const int dils[3] = {1, 2, 4};
  float* couts[3] = {cbuf0, cbuf1, cbuf2};
  for (int s = 0; s < 3; ++s) {
    ConvP cp = {hbuf, conv_w + (size_t)s * 256 * 256 * 3, conv_b + s * 256,
                bn_g + s * 256, bn_b + s * 256, bn_rm + s * 256, bn_rv + s * 256,
                couts[s], dils[s]};
    conv_wmma<<<dim3(224, 4), blk, 0, stream>>>(cp);
  }

  // softmax gate fusion -> buf1
  gate_fuse<<<dim3(3584), blk, 0, stream>>>(cbuf0, cbuf1, cbuf2, gate_w, gate_b,
                                            buf1, (int)rowsA);

  // ms-proj + residual h + PE, scattered into h2 rows 1..448; then cls row 0
  { GemmP p = {buf1, msw, nullptr, msb, nullptr,
               (int)rowsA, 256, 256, 256, 256, 256, nullptr, hbuf, h2};
    gemm_wmma<4><<<dim3(224, 4), blk, 0, stream>>>(p); }
  cls_fill<<<dim3(64), blk, 0, stream>>>(h2, cls_t);

  const size_t plane = (size_t)Bn * 8 * TT * 32;
  for (int l = 0; l < 4; ++l) {
    ln_kernel<<<dim3(3592), blk, 0, stream>>>(h2, ln1_g + l * 256, ln1_b + l * 256,
                                              buf1, (int)rowsB);
    { GemmP p = {buf1, qkv_w + (size_t)l * 256 * 768, nullptr, qkv_b + l * 768, nullptr,
                 (int)rowsB, 768, 256, 256, 768, 768, qkvh, nullptr, nullptr};
      gemm_wmma<3><<<dim3(225, 12), blk, 0, stream>>>(p); }
    attn_wmma<<<dim3(29, Bn * 8), dim3(32), 0, stream>>>(
        qkvh, qkvh + plane, qkvh + 2 * plane, alpha, l, buf1);
    { GemmP p = {buf1, out_w + (size_t)l * 256 * 256, h2, out_b + l * 256, h2,
                 (int)rowsB, 256, 256, 256, 256, 256, nullptr, nullptr, nullptr};
      gemm_wmma<2><<<dim3(225, 4), blk, 0, stream>>>(p); }
    ln_kernel<<<dim3(3592), blk, 0, stream>>>(h2, ln2_g + l * 256, ln2_b + l * 256,
                                              buf1, (int)rowsB);
    { GemmP p = {buf1, ffw1 + (size_t)l * 256 * 1024, ffnMid, ffb1 + l * 1024, nullptr,
                 (int)rowsB, 1024, 256, 256, 1024, 1024, nullptr, nullptr, nullptr};
      gemm_wmma<1><<<dim3(225, 16), blk, 0, stream>>>(p); }
    { GemmP p = {ffnMid, ffw2 + (size_t)l * 1024 * 256, h2, ffb2 + l * 256, h2,
                 (int)rowsB, 256, 1024, 1024, 256, 256, nullptr, nullptr, nullptr};
      gemm_wmma<2><<<dim3(225, 4), blk, 0, stream>>>(p); }
  }

  ln_kernel<<<dim3(3592), blk, 0, stream>>>(h2, fin_g, fin_b, buf1, (int)rowsB);
  cls_head<<<dim3(64), blk, 0, stream>>>(buf1, cw1, cb1, cw2, cb2, out);

  (void)in_sizes; (void)n_in; (void)out_size; (void)ws_size;
}
